// EnergyPredictor_v1_24859270709518
// MI455X (gfx1250) — matrix operand order repaired and checked by reading, not compiled, not run on hardware
//
#include <hip/hip_runtime.h>
#include <hip/hip_bf16.h>

typedef _Float16 v16h __attribute__((ext_vector_type(16)));
typedef _Float16 v8h  __attribute__((ext_vector_type(8)));
typedef float    v8f  __attribute__((ext_vector_type(8)));

#define Bb   8
#define Nn   2000
#define NNb  64
#define CNTF 1024000.0f   /* B*N*NN */
#define EPSBN 1e-5f
#define SIG  0.01f

// ---------------------------------------------------------------------------
// Main fused pass: features -> L0 -> L1 [-> bn1 -> L2 [-> bn2 -> L3 -> pool]]
// stage 1: emit per-WG partial (sum, sumsq) of t1 = h0@W1^T + b1
// stage 2: apply bn1, emit partials of t2
// stage 3: apply bn1, bn2, L3, neighbor mean-pool, prior term
// One workgroup (128 threads / 4 waves) per (b, n) particle; wave r owns
// neighbor rows 16r..16r+15; 16x64 f32 accumulator = 4 x v_wmma 16x16x32 f16.
// ---------------------------------------------------------------------------
__global__ __launch_bounds__(128) void ep_main_pass(
    int stage,
    const float* __restrict__ pos, const float* __restrict__ ori, const int* __restrict__ nbr,
    const float* __restrict__ w0, const float* __restrict__ b0,
    const float* __restrict__ w1, const float* __restrict__ b1,
    const float* __restrict__ g1, const float* __restrict__ be1,
    const float* __restrict__ w2, const float* __restrict__ b2,
    const float* __restrict__ g2, const float* __restrict__ be2,
    const float* __restrict__ w3, const float* __restrict__ b3,
    const float* __restrict__ stats,     // [256]: m1[64], inv1[64], m2[64], inv2[64]
    float* __restrict__ part,            // [B*N][128] partial stats
    float* __restrict__ pooled,          // [B*N][64]
    float* __restrict__ prior)           // [B*N]
{
    __shared__ __align__(32) _Float16 w0l[64*32];   // K padded 19 -> 32
    __shared__ __align__(32) _Float16 w1l[64*64];
    __shared__ __align__(32) _Float16 w2l[64*64];
    __shared__ __align__(32) _Float16 w3l[64*64];
    __shared__ __align__(16) _Float16 stag[64*64];  // activation tile (rows x ch), f16
    __shared__ float stagF[64*64];                  // final f32 tile for pooling
    __shared__ float red1[512], red2[512];
    __shared__ float priorl[64];
    __shared__ float b0l[64], b1l[64], g1l[64], be1l[64], b2l[64], g2l[64], be2l[64], b3l[64];
    __shared__ float stl[256];

    const int t = threadIdx.x;
    for (int i = t; i < 64*32; i += 128) { int n_ = i >> 5, k = i & 31;
        w0l[i] = (_Float16)(k < 19 ? w0[n_*19 + k] : 0.f); }
    for (int i = t; i < 4096; i += 128) {
        w1l[i] = (_Float16)w1[i]; w2l[i] = (_Float16)w2[i]; w3l[i] = (_Float16)w3[i]; }
    if (t < 64) { b0l[t]=b0[t]; b1l[t]=b1[t]; g1l[t]=g1[t]; be1l[t]=be1[t];
                  b2l[t]=b2[t]; g2l[t]=g2[t]; be2l[t]=be2[t]; b3l[t]=b3[t]; }
    if (stage >= 2) stl[t] = stats[t];
    if (stage == 3) stl[128 + t] = stats[128 + t];
    __syncthreads();

    const int wg = blockIdx.x;
    const int b = wg / Nn, n = wg % Nn;
    const int r = t >> 5, lane = t & 31, lrow = lane & 15, hi = lane >> 4;
    const int m = r*16 + lrow;                      // neighbor index 0..63

    // ---- features (each lane of a pair computes its row's full 19 feats) ----
    const float px = pos[(b*Nn+n)*3+0], py = pos[(b*Nn+n)*3+1], pz = pos[(b*Nn+n)*3+2];
    float Ri[9];
    #pragma unroll
    for (int i = 0; i < 9; ++i) Ri[i] = ori[(b*Nn+n)*9 + i];

    const int j = nbr[(b*Nn+n)*NNb + m];
    float dx = pos[(b*Nn+j)*3+0]-px, dy = pos[(b*Nn+j)*3+1]-py, dz = pos[(b*Nn+j)*3+2]-pz;
    dx -= rintf(dx); dy -= rintf(dy); dz -= rintf(dz);   // BOX = 1 minimum image
    float R2 = dx*dx + dy*dy + dz*dz + 1e-12f;
    float Rv = sqrtf(R2), inv = 1.f / Rv;
    float u0 = dx*inv, u1 = dy*inv, u2 = dz*inv;
    float Rj[9];
    #pragma unroll
    for (int i = 0; i < 9; ++i) Rj[i] = ori[(b*Nn+j)*9 + i];

    float ft[19];
    ft[0]=u0; ft[1]=u1; ft[2]=u2; ft[3]=Rv;
    #pragma unroll
    for (int c = 0; c < 3; ++c) ft[4+c] = u0*Ri[c] + u1*Ri[3+c] + u2*Ri[6+c];
    #pragma unroll
    for (int c = 0; c < 3; ++c) ft[7+c] = u0*Rj[c] + u1*Rj[3+c] + u2*Rj[6+c];
    #pragma unroll
    for (int jc = 0; jc < 3; ++jc)
        #pragma unroll
        for (int kc = 0; kc < 3; ++kc)
            ft[10 + jc*3 + kc] = Ri[jc]*Rj[kc] + Ri[3+jc]*Rj[3+kc] + Ri[6+jc]*Rj[6+kc];

    if (stage == 3 && hi == 0) {
        float s1 = SIG * inv, s2 = s1*s1, s6 = s2*s2*s2;
        priorl[m] = s6*s6;                           // (sigma/R)^12
    }

    // ---- pack A fragment for layer-0 (16x32, K padded) ----
    // lane<16: halves 0..7 = K0..7, halves 8..15 = K16..23
    // lane>=16: halves 0..7 = K8..15, halves 8..15 = K24..31
    v16h a0;
    if (!hi) {
        #pragma unroll
        for (int i = 0; i < 8; ++i) a0[i] = (_Float16)ft[i];
        a0[8]=(_Float16)ft[16]; a0[9]=(_Float16)ft[17]; a0[10]=(_Float16)ft[18];
        #pragma unroll
        for (int i = 11; i < 16; ++i) a0[i] = (_Float16)0.f;
    } else {
        #pragma unroll
        for (int i = 0; i < 8; ++i) a0[i] = (_Float16)ft[8+i];
        #pragma unroll
        for (int i = 8; i < 16; ++i) a0[i] = (_Float16)0.f;
    }

    const int rowoff = (r*16 + lrow)*64;

    // K=64 GEMM from staged f16 activations: acc = stag_rows @ W^T
    auto gemm64 = [&](const _Float16* wl, v8f* ac) {
        #pragma unroll
        for (int nt = 0; nt < 4; ++nt) { v8f z = {}; ac[nt] = z; }
        #pragma unroll
        for (int kc = 0; kc < 2; ++kc) {
            v8h lo  = *(const v8h*)(stag + rowoff + kc*32 + hi*8);
            v8h hi8 = *(const v8h*)(stag + rowoff + kc*32 + 16 + hi*8);
            v16h a;
            #pragma unroll
            for (int i = 0; i < 8; ++i) { a[i] = lo[i]; a[8+i] = hi8[i]; }
            #pragma unroll
            for (int nt = 0; nt < 4; ++nt) {
                v8h b0v = *(const v8h*)(wl + (nt*16 + lrow)*64 + kc*32 + hi*8);
                v8h b1v = *(const v8h*)(wl + (nt*16 + lrow)*64 + kc*32 + 16 + hi*8);
                v16h bf;
                #pragma unroll
                for (int i = 0; i < 8; ++i) { bf[i] = b0v[i]; bf[8+i] = b1v[i]; }
                ac[nt] = __builtin_amdgcn_wmma_f32_16x16x32_f16(
                    false, a, false, bf, (short)0, ac[nt], false, false);
            }
        }
    };

    // ---- layer 0: 19 -> 64, relu ----
    v8f acc[4];
    #pragma unroll
    for (int nt = 0; nt < 4; ++nt) {
        v8h b0v = *(const v8h*)(w0l + (nt*16 + lrow)*32 + hi*8);
        v8h b1v = *(const v8h*)(w0l + (nt*16 + lrow)*32 + 16 + hi*8);
        v16h bf;
        #pragma unroll
        for (int i = 0; i < 8; ++i) { bf[i] = b0v[i]; bf[8+i] = b1v[i]; }
        v8f z = {};
        acc[nt] = __builtin_amdgcn_wmma_f32_16x16x32_f16(
            false, a0, false, bf, (short)0, z, false, false);
    }
    #pragma unroll
    for (int nt = 0; nt < 4; ++nt) {
        int ch = nt*16 + lrow; float bv = b0l[ch];
        #pragma unroll
        for (int v = 0; v < 8; ++v) {
            float x = acc[nt][v] + bv; x = x > 0.f ? x : 0.f;
            int row = r*16 + v + hi*8;
            stag[row*64 + ch] = (_Float16)x;
        }
    }

    // ---- layer 1 pre-activation ----
    gemm64(w1l, acc);

    if (stage == 1) {
        #pragma unroll
        for (int nt = 0; nt < 4; ++nt) {
            int ch = nt*16 + lrow; float bv = b1l[ch], s = 0.f, q = 0.f;
            #pragma unroll
            for (int v = 0; v < 8; ++v) { float x = acc[nt][v] + bv; s += x; q += x*x; }
            red1[nt*128 + r*32 + lane] = s; red2[nt*128 + r*32 + lane] = q;
        }
        __syncthreads();
        if (t < 64) {
            int nt = t >> 4, l = t & 15; float s = 0.f, q = 0.f;
            #pragma unroll
            for (int rr = 0; rr < 4; ++rr) {
                s += red1[nt*128 + rr*32 + l] + red1[nt*128 + rr*32 + 16 + l];
                q += red2[nt*128 + rr*32 + l] + red2[nt*128 + rr*32 + 16 + l];
            }
            part[wg*128 + t] = s; part[wg*128 + 64 + t] = q;
        }
        return;
    }

    // ---- bn1 + relu ----
    #pragma unroll
    for (int nt = 0; nt < 4; ++nt) {
        int ch = nt*16 + lrow;
        float mean = stl[ch], rv = stl[64+ch], g = g1l[ch], bt = be1l[ch], bv = b1l[ch];
        #pragma unroll
        for (int v = 0; v < 8; ++v) {
            float x = acc[nt][v] + bv;
            x = (x - mean)*rv*g + bt; x = x > 0.f ? x : 0.f;
            int row = r*16 + v + hi*8;
            stag[row*64 + ch] = (_Float16)x;
        }
    }

    // ---- layer 2 pre-activation ----
    gemm64(w2l, acc);

    if (stage == 2) {
        #pragma unroll
        for (int nt = 0; nt < 4; ++nt) {
            int ch = nt*16 + lrow; float bv = b2l[ch], s = 0.f, q = 0.f;
            #pragma unroll
            for (int v = 0; v < 8; ++v) { float x = acc[nt][v] + bv; s += x; q += x*x; }
            red1[nt*128 + r*32 + lane] = s; red2[nt*128 + r*32 + lane] = q;
        }
        __syncthreads();
        if (t < 64) {
            int nt = t >> 4, l = t & 15; float s = 0.f, q = 0.f;
            #pragma unroll
            for (int rr = 0; rr < 4; ++rr) {
                s += red1[nt*128 + rr*32 + l] + red1[nt*128 + rr*32 + 16 + l];
                q += red2[nt*128 + rr*32 + l] + red2[nt*128 + rr*32 + 16 + l];
            }
            part[wg*128 + t] = s; part[wg*128 + 64 + t] = q;
        }
        return;
    }

    // ---- stage 3: bn2 + relu, layer 3, pool ----
    #pragma unroll
    for (int nt = 0; nt < 4; ++nt) {
        int ch = nt*16 + lrow;
        float mean = stl[128+ch], rv = stl[192+ch], g = g2l[ch], bt = be2l[ch], bv = b2l[ch];
        #pragma unroll
        for (int v = 0; v < 8; ++v) {
            float x = acc[nt][v] + bv;
            x = (x - mean)*rv*g + bt; x = x > 0.f ? x : 0.f;
            int row = r*16 + v + hi*8;
            stag[row*64 + ch] = (_Float16)x;
        }
    }
    gemm64(w3l, acc);
    #pragma unroll
    for (int nt = 0; nt < 4; ++nt) {
        int ch = nt*16 + lrow; float bv = b3l[ch];
        #pragma unroll
        for (int v = 0; v < 8; ++v) {
            int row = r*16 + v + hi*8;
            stagF[row*64 + ch] = acc[nt][v] + bv;
        }
    }
    __syncthreads();
    if (t < 64) {                                   // mean over 64 neighbors, fixed order
        float s = 0.f;
        for (int row = 0; row < 64; ++row) s += stagF[row*64 + t];
        pooled[wg*64 + t] = s * (1.f/64.f);
    }
    if (t == 0) {
        float s = 0.f;
        for (int i = 0; i < 64; ++i) s += priorl[i];
        prior[wg] = s;
    }
}

// ---------------------------------------------------------------------------
__global__ __launch_bounds__(128) void ep_reduce_stats(
    const float* __restrict__ part, int nwg, float* __restrict__ out)
{
    int t = threadIdx.x;
    float s = 0.f;
    for (int i = 0; i < nwg; ++i) s += part[i*128 + t];
    __shared__ float sh[128];
    sh[t] = s; __syncthreads();
    if (t < 64) {
        float mean = sh[t] / CNTF;
        float var  = sh[64 + t] / CNTF - mean*mean;
        out[t] = mean;
        out[64 + t] = 1.f / sqrtf(var + EPSBN);
    }
}

// per-batch per-channel max over N rows
__global__ __launch_bounds__(256) void ep_colmax(
    const float* __restrict__ x, float* __restrict__ mx)
{
    int b = blockIdx.x, t = threadIdx.x, c = t & 63, g = t >> 6;
    float m = -3.0e38f;
    for (int n = g; n < Nn; n += 4) m = fmaxf(m, x[(b*Nn + n)*64 + c]);
    __shared__ float sh[256];
    sh[t] = m; __syncthreads();
    if (t < 64) mx[b*64 + t] = fmaxf(fmaxf(sh[t], sh[64+t]), fmaxf(sh[128+t], sh[192+t]));
}

// z_out = tanh((x - colmax) @ w^T + b); 4 rows x 64 channels per block
__global__ __launch_bounds__(256) void ep_pe(
    const float* __restrict__ x, const float* __restrict__ mx,
    const float* __restrict__ w, const float* __restrict__ bias,
    float* __restrict__ out)
{
    int bb = blockIdx.x / 500, rb = blockIdx.x % 500;
    int t = threadIdx.x, c = t & 63, rr = t >> 6;
    __shared__ float wl[4096], xl[4][64], mxl[64];
    for (int i = t; i < 4096; i += 256) wl[i] = w[i];
    if (t < 64) mxl[t] = mx[bb*64 + t];
    int row = rb*4 + rr;
    xl[rr][c] = x[(bb*Nn + row)*64 + c];
    __syncthreads();
    float a = bias[c];
    for (int k = 0; k < 64; ++k) a += (xl[rr][k] - mxl[k]) * wl[c*64 + k];
    out[(bb*Nn + row)*64 + c] = tanhf(a);
}

__global__ __launch_bounds__(256) void ep_final(
    const float* __restrict__ s, const float* __restrict__ r1w,
    const float* __restrict__ r1b, const float* __restrict__ r2w,
    const float* __restrict__ r2b, const float* __restrict__ prior,
    float* __restrict__ out)
{
    int b = blockIdx.x, t = threadIdx.x;
    __shared__ float sl[64], hl[64], pr[256];
    if (t < 64) sl[t] = s[b*64 + t];
    __syncthreads();
    if (t < 64) {
        float a = r1b[t];
        for (int k = 0; k < 64; ++k) a += sl[k] * r1w[t*64 + k];
        hl[t] = tanhf(a);
    }
    float p = 0.f;
    for (int n = t; n < Nn; n += 256) p += prior[b*Nn + n];
    pr[t] = p; __syncthreads();
    for (int st = 128; st > 0; st >>= 1) { if (t < st) pr[t] += pr[t + st]; __syncthreads(); }
    if (t == 0) {
        float e = r2b[0];
        for (int k = 0; k < 64; ++k) e += hl[k] * r2w[k];
        out[b] = e + pr[0];
    }
}

// ---------------------------------------------------------------------------
extern "C" void kernel_launch(void* const* d_in, const int* in_sizes, int n_in,
                              void* d_out, int out_size, void* d_ws, size_t ws_size,
                              hipStream_t stream) {
    (void)in_sizes; (void)n_in; (void)out_size; (void)ws_size;
    const float* pos = (const float*)d_in[0];
    const float* ori = (const float*)d_in[1];
    const int*   nbr = (const int*)  d_in[2];
    const float* w0  = (const float*)d_in[3];  const float* b0  = (const float*)d_in[4];
    const float* w1  = (const float*)d_in[5];  const float* b1  = (const float*)d_in[6];
    const float* g1  = (const float*)d_in[7];  const float* be1 = (const float*)d_in[8];
    const float* w2  = (const float*)d_in[9];  const float* b2  = (const float*)d_in[10];
    const float* g2  = (const float*)d_in[11]; const float* be2 = (const float*)d_in[12];
    const float* w3  = (const float*)d_in[13]; const float* b3  = (const float*)d_in[14];
    const float* p1w = (const float*)d_in[15]; const float* p1b = (const float*)d_in[16];
    const float* p2w = (const float*)d_in[17]; const float* p2b = (const float*)d_in[18];
    const float* p3w = (const float*)d_in[19]; const float* p3b = (const float*)d_in[20];
    const float* r1w = (const float*)d_in[21]; const float* r1b = (const float*)d_in[22];
    const float* r2w = (const float*)d_in[23]; const float* r2b = (const float*)d_in[24];

    // workspace layout (bytes):
    //   [0, 8192000)        : partial stats (16000 x 128 f32); later reused as zA/zB
    //   [8192000, 12288000) : pooled (16000 x 64 f32)
    //   [12288000,12352000) : prior (16000 f32)
    //   [12352000,12353024) : stats (256 f32)
    //   [12353024,12355072) : colmax / s vectors (8 x 64 f32)
    char*  ws     = (char*)d_ws;
    float* part   = (float*)ws;
    float* zA     = (float*)ws;
    float* zB     = (float*)(ws + 4096000);
    float* pooled = (float*)(ws + 8192000);
    float* prior  = (float*)(ws + 12288000);
    float* stats  = (float*)(ws + 12352000);
    float* mx     = (float*)(ws + 12353024);

    dim3 big(Bb*Nn), tb(128);
    ep_main_pass<<<big, tb, 0, stream>>>(1, pos,ori,nbr, w0,b0,w1,b1,g1,be1,
                                         w2,b2,g2,be2,w3,b3, stats, part,pooled,prior);
    ep_reduce_stats<<<1, 128, 0, stream>>>(part, Bb*Nn, stats);
    ep_main_pass<<<big, tb, 0, stream>>>(2, pos,ori,nbr, w0,b0,w1,b1,g1,be1,
                                         w2,b2,g2,be2,w3,b3, stats, part,pooled,prior);
    ep_reduce_stats<<<1, 128, 0, stream>>>(part, Bb*Nn, stats + 128);
    ep_main_pass<<<big, tb, 0, stream>>>(3, pos,ori,nbr, w0,b0,w1,b1,g1,be1,
                                         w2,b2,g2,be2,w3,b3, stats, part,pooled,prior);

    // DTanh tail: pooled -> z1 -> z2 -> z3 -> max -> head (+ prior)
    ep_colmax<<<8, 256, 0, stream>>>(pooled, mx);
    ep_pe<<<dim3(8*500), 256, 0, stream>>>(pooled, mx, p1w, p1b, zA);
    ep_colmax<<<8, 256, 0, stream>>>(zA, mx);
    ep_pe<<<dim3(8*500), 256, 0, stream>>>(zA, mx, p2w, p2b, zB);
    ep_colmax<<<8, 256, 0, stream>>>(zB, mx);
    ep_pe<<<dim3(8*500), 256, 0, stream>>>(zB, mx, p3w, p3b, zA);
    ep_colmax<<<8, 256, 0, stream>>>(zA, mx);               // s = z3.max(1)
    ep_final<<<8, 256, 0, stream>>>(mx, r1w, r1b, r2w, r2b, prior, (float*)d_out);
}
